// Encoder_43963285241949
// MI455X (gfx1250) — matrix operand order repaired and checked by reading, compiled, not run on hardware
//
#include <hip/hip_runtime.h>

// ---------------------------------------------------------------------------
// CDNA5 (gfx1250) wave32 WMMA transformer encoder layer.
// - All GEMMs: v_wmma_f32_16x16x32_bf16 (bf16 A/B, f32 accumulate)
// - Global->LDS staging of A-tiles / attention K-tiles uses the CDNA5 async
//   path: global_load_async_to_lds_b128 (+ s_wait_asynccnt), ISA ch10.7
// - GEMM double-buffers LDS so async loads overlap the WMMA stream
// - Attention is flash-style (streaming softmax): scores never touch HBM
// ---------------------------------------------------------------------------

typedef __bf16 bf16x16 __attribute__((ext_vector_type(16)));
typedef float f32x8 __attribute__((ext_vector_type(8)));

union Frag {
    unsigned u[8];
    bf16x16 v;
};

__device__ __forceinline__ f32x8 zero8() {
    f32x8 z = {0.f, 0.f, 0.f, 0.f, 0.f, 0.f, 0.f, 0.f};
    return z;
}

// round-to-nearest-even f32 -> bf16
__device__ __forceinline__ unsigned short f2bf(float f) {
    unsigned x = __float_as_uint(f);
    unsigned r = x + 0x7FFFu + ((x >> 16) & 1u);
    return (unsigned short)(r >> 16);
}

// Async DMA of 16 bytes global -> LDS. VDST holds the LDS byte offset
// (low 32 bits of the generic address of a __shared__ object), VADDR is the
// 64-bit global address. Tracked by ASYNCcnt.
__device__ __forceinline__ void async_ld_b128(unsigned lds_off, const void* gptr) {
    asm volatile("global_load_async_to_lds_b128 %0, %1, off"
                 :
                 : "v"(lds_off), "v"((unsigned long long)(uintptr_t)gptr)
                 : "memory");
}
__device__ __forceinline__ void wait_async0() {
    asm volatile("s_wait_asynccnt 0x0" ::: "memory");
}
__device__ __forceinline__ unsigned lds_off_of(const void* p) {
    return (unsigned)(uintptr_t)p;  // generic LDS addr: low 32 bits = LDS offset
}

// A-matrix 16x32 bf16 fragment from LDS, row-major tile with leading dim ldk.
// ISA 7.12.2: lane holds row lane&15; reg v holds K pair
//   k = kk + (v>>2)*16 + (lane>>4)*8 + (v&3)*2   (pair k,k+1 packed in 32b)
__device__ __forceinline__ Frag load_a_frag(const unsigned short* lds, int rowBase,
                                            int ldk, int kk, int lane) {
    Frag f;
    const int r = rowBase + (lane & 15);
    const int hf = lane >> 4;
#pragma unroll
    for (int v = 0; v < 8; ++v) {
        int k = kk + ((v >> 2) << 4) + (hf << 3) + ((v & 3) << 1);
        f.u[v] = *(const unsigned*)&lds[r * ldk + k];
    }
    return f;
}

// B-matrix 32x16 bf16 fragment from LDS stored TRANSPOSED: Bt[n][k], ld ldk.
// Lane holds col n = lane&15; reg v holds pair k = (v>>2)*16 + (lane>>4)*8 + (v&3)*2.
__device__ __forceinline__ Frag load_b_frag(const unsigned short* ldsT, int nBase,
                                            int ldk, int kk, int lane) {
    Frag f;
    const int n = nBase + (lane & 15);
    const int kb = kk + ((lane >> 4) << 3);
#pragma unroll
    for (int v = 0; v < 8; ++v) {
        f.u[v] = *(const unsigned*)&ldsT[n * ldk + kb + ((v >> 2) << 4) + ((v & 3) << 1)];
    }
    return f;
}

__device__ __forceinline__ f32x8 wmma_bf16(const Frag& a, const Frag& b, f32x8 c) {
    return __builtin_amdgcn_wmma_f32_16x16x32_bf16(false, a.v, false, b.v,
                                                   (short)0, c, false, false);
}

// ---------------------------------------------------------------------------
// Generic bf16 GEMM: C[M,N] = A[M,K] @ B[K,N]  (+bias, +relu, f32/bf16 out)
// Block tile 128x128, 8 waves as 4(M) x 2(N), wave tile 32x64 (2x4 WMMA tiles).
// K staged through double-buffered LDS in chunks of 64 (two wmma K-steps):
//   A chunk: async global->LDS DMA (no VGPR round-trip)
//   B chunk: global->regs prefetch, transposed ds_store after compute
// ---------------------------------------------------------------------------
template <bool BF16OUT, bool RELU, bool BIAS>
__global__ __launch_bounds__(256) void gemm_kernel(
    const unsigned short* __restrict__ A, const unsigned short* __restrict__ B,
    const float* __restrict__ bias, float* __restrict__ Cf,
    unsigned short* __restrict__ Cb, int M, int N, int K) {
    __shared__ unsigned short As[2][128 * 64];  // [m][k]
    __shared__ unsigned short Bs[2][128 * 64];  // transposed: [n][k]
    const int tid = threadIdx.x;
    const int lane = tid & 31;
    const int wave = tid >> 5;
    const int waveM = wave >> 1;
    const int waveN = wave & 1;
    const int m0 = blockIdx.y * 128;
    const int n0 = blockIdx.x * 128;
    const int nChunks = K >> 6;

    f32x8 acc[2][4];
#pragma unroll
    for (int i = 0; i < 2; ++i)
#pragma unroll
        for (int j = 0; j < 4; ++j) acc[i][j] = zero8();

    uint4 breg[4];

    auto issueA = [&](int k0, int buf) {
#pragma unroll
        for (int i = 0; i < 4; ++i) {
            int vec = tid + i * 256;           // 1024 16B vectors of the A tile
            int r = vec >> 3;
            int c = (vec & 7) << 3;
            async_ld_b128(lds_off_of(&As[buf][r * 64 + c]),
                          &A[(size_t)(m0 + r) * K + k0 + c]);
        }
    };
    auto loadBregs = [&](int k0) {
#pragma unroll
        for (int i = 0; i < 4; ++i) {
            int vec = tid + i * 256;
            int kr = vec >> 4;
            int nc = (vec & 15) << 3;
            breg[i] = *(const uint4*)&B[(size_t)(k0 + kr) * N + n0 + nc];
        }
    };
    auto storeB = [&](int buf) {
#pragma unroll
        for (int i = 0; i < 4; ++i) {
            int vec = tid + i * 256;
            int kr = vec >> 4;
            int nc = (vec & 15) << 3;
            const unsigned short* ds = (const unsigned short*)&breg[i];
#pragma unroll
            for (int j = 0; j < 8; ++j) Bs[buf][(nc + j) * 64 + kr] = ds[j];
        }
    };

    // Pipeline prologue: chunk 0 into buffer 0
    issueA(0, 0);
    loadBregs(0);
    storeB(0);
    wait_async0();
    __syncthreads();

    for (int c = 0; c < nChunks; ++c) {
        const int cur = c & 1;
        if (c + 1 < nChunks) {
            issueA((c + 1) << 6, cur ^ 1);   // DMA next A chunk into other buffer
            loadBregs((c + 1) << 6);         // next B chunk in flight in VGPRs
        }
        // Compute on current buffer: 16 WMMAs
#pragma unroll
        for (int kk = 0; kk < 64; kk += 32) {
            Frag a0 = load_a_frag(As[cur], waveM * 32, 64, kk, lane);
            Frag a1 = load_a_frag(As[cur], waveM * 32 + 16, 64, kk, lane);
#pragma unroll
            for (int t = 0; t < 4; ++t) {
                Frag b = load_b_frag(Bs[cur], waveN * 64 + t * 16, 64, kk, lane);
                acc[0][t] = wmma_bf16(a0, b, acc[0][t]);
                acc[1][t] = wmma_bf16(a1, b, acc[1][t]);
            }
        }
        if (c + 1 < nChunks) {
            storeB(cur ^ 1);
            wait_async0();   // our async DMAs into cur^1 are done
            __syncthreads(); // everyone's stores visible before next compute
        }
    }

    // Epilogue. C layout: lane holds col lane&15; reg v is row (lane>>4)*8 + v.
    const int hf = lane >> 4;
    const int nl = lane & 15;
#pragma unroll
    for (int mt = 0; mt < 2; ++mt) {
#pragma unroll
        for (int t = 0; t < 4; ++t) {
            int col = n0 + waveN * 64 + t * 16 + nl;
            float bv = BIAS ? bias[col] : 0.0f;
#pragma unroll
            for (int v = 0; v < 8; ++v) {
                int row = m0 + waveM * 32 + mt * 16 + hf * 8 + v;
                float x = acc[mt][t][v] + bv;
                if (RELU) x = fmaxf(x, 0.0f);
                if (BF16OUT)
                    Cb[(size_t)row * N + col] = f2bf(x);
                else
                    Cf[(size_t)row * N + col] = x;
            }
        }
    }
}

// ---------------------------------------------------------------------------
// Flash attention. QKV packed bf16 [token][3072] (Q|K|V, head-concat inside).
// Block: one (b,h), 64 query rows; 4 waves x 16 rows. K/V chunks of 64 in LDS:
// K chunk via async global->LDS DMA, V chunk transposed through VGPRs.
// Output written bf16 directly into head-concat layout [token][1024].
// ---------------------------------------------------------------------------
__global__ __launch_bounds__(128) void attention_kernel(
    const unsigned short* __restrict__ QKV, unsigned short* __restrict__ O) {
    __shared__ unsigned short Ks[64 * 64];      // [s][e] == Bt for Q@K^T (n=s,k=e)
    __shared__ unsigned short Vt[64 * 64];      // [e][s] == Bt for P@V   (n=e,k=s)
    __shared__ unsigned short Ps[4][16 * 64];   // per-wave P staging (A layout src)
    const int tid = threadIdx.x;
    const int lane = tid & 31;
    const int wave = tid >> 5;
    const int hf = lane >> 4;
    const int nl = lane & 15;
    const int bh = blockIdx.y;
    const int b = bh >> 4;
    const int h = bh & 15;
    const int rowTile = blockIdx.x * 64;
    const size_t tokenBase = (size_t)b * 2048;
    const float scale = 0.03125f;  // 1/sqrt(d_model) = 1/32 (reference scaling)

    // Q fragments for this wave's 16 rows (K-dim = head_size 64 -> 2 steps)
    Frag qf[2];
    {
        int r = rowTile + wave * 16 + nl;
        size_t base = (tokenBase + r) * 3072 + h * 64;
#pragma unroll
        for (int kkI = 0; kkI < 2; ++kkI) {
#pragma unroll
            for (int v = 0; v < 8; ++v) {
                int k = kkI * 32 + ((v >> 2) << 4) + (hf << 3) + ((v & 3) << 1);
                qf[kkI].u[v] = *(const unsigned*)&QKV[base + k];
            }
        }
    }

    f32x8 Oacc[4];
#pragma unroll
    for (int t = 0; t < 4; ++t) Oacc[t] = zero8();
    float mrun[8], lrun[8];
#pragma unroll
    for (int v = 0; v < 8; ++v) { mrun[v] = -1e30f; lrun[v] = 0.0f; }

    for (int sc = 0; sc < 2048; sc += 64) {
        __syncthreads();  // prior chunk fully consumed before overwrite
        // K chunk [64 s][64 e] via async DMA (row-contiguous 16B pieces)
#pragma unroll
        for (int i = 0; i < 4; ++i) {
            int vec = tid + i * 128;
            int s = vec >> 3;
            int e = (vec & 7) << 3;
            async_ld_b128(lds_off_of(&Ks[s * 64 + e]),
                          &QKV[(tokenBase + sc + s) * 3072 + 1024 + h * 64 + e]);
        }
        // V chunk transposed [e][s] (transpose needs the VGPR round-trip)
#pragma unroll
        for (int i = 0; i < 16; ++i) {
            int vec = tid + i * 128;
            int s = vec >> 5;
            int e = (vec & 31) << 1;
            unsigned d = *(const unsigned*)&QKV[(tokenBase + sc + s) * 3072 + 2048 + h * 64 + e];
            Vt[e * 64 + s] = (unsigned short)(d & 0xFFFFu);
            Vt[(e + 1) * 64 + s] = (unsigned short)(d >> 16);
        }
        wait_async0();
        __syncthreads();

        // Scores S = Q @ K^T  (16 rows x 64 cols per wave)
        f32x8 Sacc[4];
#pragma unroll
        for (int t = 0; t < 4; ++t) Sacc[t] = zero8();
#pragma unroll
        for (int kkI = 0; kkI < 2; ++kkI) {
#pragma unroll
            for (int t = 0; t < 4; ++t) {
                Frag bk = load_b_frag(Ks, t * 16, 64, kkI * 32, lane);
                Sacc[t] = wmma_bf16(qf[kkI], bk, Sacc[t]);
            }
        }

        // Streaming softmax. Row m = hf*8 + v lives on the 16 lanes of one half,
        // so butterflies over xor masks 1,2,4,8 reduce exactly one row.
#pragma unroll
        for (int v = 0; v < 8; ++v) {
            float mx = fmaxf(fmaxf(Sacc[0][v], Sacc[1][v]), fmaxf(Sacc[2][v], Sacc[3][v]));
#pragma unroll
            for (int off = 8; off >= 1; off >>= 1) mx = fmaxf(mx, __shfl_xor(mx, off, 32));
            mx *= scale;
            float mnew = fmaxf(mrun[v], mx);
            float alpha = __expf(mrun[v] - mnew);
            float p[4];
            float rs = 0.0f;
#pragma unroll
            for (int t = 0; t < 4; ++t) {
                p[t] = __expf(Sacc[t][v] * scale - mnew);
                rs += p[t];
            }
#pragma unroll
            for (int off = 8; off >= 1; off >>= 1) rs += __shfl_xor(rs, off, 32);
            lrun[v] = lrun[v] * alpha + rs;
            mrun[v] = mnew;
#pragma unroll
            for (int t = 0; t < 4; ++t) Oacc[t][v] *= alpha;
            int m = hf * 8 + v;
#pragma unroll
            for (int t = 0; t < 4; ++t) Ps[wave][m * 64 + t * 16 + nl] = f2bf(p[t]);
        }

        // O += P @ V  (per-wave P region, same-wave DS ops are in-order)
#pragma unroll
        for (int kkI = 0; kkI < 2; ++kkI) {
            Frag pa = load_a_frag(Ps[wave], 0, 64, kkI * 32, lane);
#pragma unroll
            for (int t = 0; t < 4; ++t) {
                Frag bv = load_b_frag(Vt, t * 16, 64, kkI * 32, lane);
                Oacc[t] = wmma_bf16(pa, bv, Oacc[t]);
            }
        }
    }

    // Normalize and write into head-concat layout.
#pragma unroll
    for (int v = 0; v < 8; ++v) {
        float inv = 1.0f / lrun[v];
        int r = rowTile + wave * 16 + hf * 8 + v;
        size_t base = (tokenBase + r) * 1024 + h * 64;
#pragma unroll
        for (int t = 0; t < 4; ++t) O[base + t * 16 + nl] = f2bf(Oacc[t][v] * inv);
    }
}

// ---------------------------------------------------------------------------
// Residual add + reference's LayerNorm variant:
//   y = g * ((v + mean) / (std_unbiased + eps)) + beta,  v = a + r
// One block (256 thr) per token of 1024 features.
// ---------------------------------------------------------------------------
__global__ __launch_bounds__(256) void add_ln_kernel(
    const float* __restrict__ a, const float* __restrict__ r,
    const float* __restrict__ g, const float* __restrict__ beta,
    float* __restrict__ outf, unsigned short* __restrict__ outb) {
    __shared__ float s1[8], s2[8];
    const int tid = threadIdx.x;
    const size_t base = (size_t)blockIdx.x * 1024;
    float vv[4];
    float sum = 0.0f, sq = 0.0f;
#pragma unroll
    for (int j = 0; j < 4; ++j) {
        int i = tid + j * 256;
        vv[j] = a[base + i] + r[base + i];
        sum += vv[j];
        sq += vv[j] * vv[j];
    }
#pragma unroll
    for (int off = 16; off >= 1; off >>= 1) {
        sum += __shfl_xor(sum, off, 32);
        sq += __shfl_xor(sq, off, 32);
    }
    if ((tid & 31) == 0) { s1[tid >> 5] = sum; s2[tid >> 5] = sq; }
    __syncthreads();
    float S = 0.0f, SQ = 0.0f;
#pragma unroll
    for (int w = 0; w < 8; ++w) { S += s1[w]; SQ += s2[w]; }
    float mean = S * (1.0f / 1024.0f);
    float var = (SQ - 1024.0f * mean * mean) * (1.0f / 1023.0f);  // ddof=1
    float denom = 1.0f / (sqrtf(fmaxf(var, 0.0f)) + 1e-6f);
#pragma unroll
    for (int j = 0; j < 4; ++j) {
        int i = tid + j * 256;
        float y = g[i] * ((vv[j] + mean) * denom) + beta[i];
        outf[base + i] = y;
        if (outb) outb[base + i] = f2bf(y);
    }
}

__global__ void cast_bf16_kernel(const float* __restrict__ in,
                                 unsigned short* __restrict__ out, int n) {
    int i = blockIdx.x * blockDim.x + threadIdx.x;
    if (i < n) out[i] = f2bf(in[i]);
}

// Wq/Wk/Wv [16][1024][64] f32 -> fused bf16 [1024 d][3072 = (q|k|v) x (h*64+e)]
__global__ void pack_qkv_kernel(const float* __restrict__ Wq,
                                const float* __restrict__ Wk,
                                const float* __restrict__ Wv,
                                unsigned short* __restrict__ out) {
    int i = blockIdx.x * blockDim.x + threadIdx.x;
    if (i >= 16 * 1024 * 64) return;
    int e = i & 63;
    int d = (i >> 6) & 1023;
    int h = i >> 16;
    size_t o = (size_t)d * 3072 + h * 64 + e;
    out[o] = f2bf(Wq[i]);
    out[o + 1024] = f2bf(Wk[i]);
    out[o + 2048] = f2bf(Wv[i]);
}

extern "C" void kernel_launch(void* const* d_in, const int* in_sizes, int n_in,
                              void* d_out, int out_size, void* d_ws, size_t ws_size,
                              hipStream_t stream) {
    (void)in_sizes; (void)n_in; (void)out_size; (void)ws_size;
    const float* x   = (const float*)d_in[0];
    const float* Wq  = (const float*)d_in[1];
    const float* Wk  = (const float*)d_in[2];
    const float* Wv  = (const float*)d_in[3];
    const float* Wo  = (const float*)d_in[4];
    const float* bo  = (const float*)d_in[5];
    const float* W1  = (const float*)d_in[6];
    const float* b1  = (const float*)d_in[7];
    const float* W2  = (const float*)d_in[8];
    const float* b2  = (const float*)d_in[9];
    const float* g1  = (const float*)d_in[10];
    const float* be1 = (const float*)d_in[11];
    const float* g2  = (const float*)d_in[12];
    const float* be2 = (const float*)d_in[13];
    float* out = (float*)d_out;

    const int T = 8192;  // tokens = 4 * 2048
    char* ws = (char*)d_ws;
    size_t off = 0;
    auto alloc = [&](size_t bytes) -> void* {
        void* p = ws + off;
        off = (off + bytes + 255) & ~(size_t)255;
        return p;
    };
    unsigned short* xbf   = (unsigned short*)alloc((size_t)T * 1024 * 2);
    unsigned short* Wqkv  = (unsigned short*)alloc((size_t)1024 * 3072 * 2);
    unsigned short* Wobf  = (unsigned short*)alloc((size_t)1024 * 1024 * 2);
    unsigned short* W1bf  = (unsigned short*)alloc((size_t)1024 * 4096 * 2);
    unsigned short* W2bf  = (unsigned short*)alloc((size_t)4096 * 1024 * 2);
    unsigned short* QKVbf = (unsigned short*)alloc((size_t)T * 3072 * 2);
    unsigned short* Obf   = (unsigned short*)alloc((size_t)T * 1024 * 2);
    float*          mha   = (float*)alloc((size_t)T * 1024 * 4);
    float*          h1    = (float*)alloc((size_t)T * 1024 * 4);
    unsigned short* h1bf  = (unsigned short*)alloc((size_t)T * 1024 * 2);
    unsigned short* ff1   = (unsigned short*)alloc((size_t)T * 4096 * 2);
    float*          ff2   = (float*)alloc((size_t)T * 1024 * 4);

    // One-time precision casts / weight packing
    cast_bf16_kernel<<<(T * 1024 + 255) / 256, 256, 0, stream>>>(x, xbf, T * 1024);
    cast_bf16_kernel<<<(1024 * 1024 + 255) / 256, 256, 0, stream>>>(Wo, Wobf, 1024 * 1024);
    cast_bf16_kernel<<<(1024 * 4096 + 255) / 256, 256, 0, stream>>>(W1, W1bf, 1024 * 4096);
    cast_bf16_kernel<<<(4096 * 1024 + 255) / 256, 256, 0, stream>>>(W2, W2bf, 4096 * 1024);
    pack_qkv_kernel<<<(16 * 1024 * 64 + 255) / 256, 256, 0, stream>>>(Wq, Wk, Wv, Wqkv);

    // Fused QKV projection: [8192,1024] @ [1024,3072] -> bf16
    gemm_kernel<true, false, false><<<dim3(3072 / 128, T / 128), 256, 0, stream>>>(
        xbf, Wqkv, nullptr, nullptr, QKVbf, T, 3072, 1024);

    // Flash attention, all 64 (b,h) pairs x 32 row-tiles
    attention_kernel<<<dim3(2048 / 64, 64), 128, 0, stream>>>(QKVbf, Obf);

    // Output projection + bias -> f32
    gemm_kernel<false, false, true><<<dim3(1024 / 128, T / 128), 256, 0, stream>>>(
        Obf, Wobf, bo, mha, nullptr, T, 1024, 1024);

    // Residual + LN1 -> h1 (f32 for next residual) + h1bf (GEMM input)
    add_ln_kernel<<<T, 256, 0, stream>>>(x, mha, g1, be1, h1, h1bf);

    // FFN1 + bias + relu -> bf16
    gemm_kernel<true, true, true><<<dim3(4096 / 128, T / 128), 256, 0, stream>>>(
        h1bf, W1bf, b1, nullptr, ff1, T, 4096, 1024);

    // FFN2 + bias -> f32
    gemm_kernel<false, false, true><<<dim3(1024 / 128, T / 128), 256, 0, stream>>>(
        ff1, W2bf, b2, ff2, nullptr, T, 1024, 4096);

    // Residual + LN2 -> final output
    add_ln_kernel<<<T, 256, 0, stream>>>(h1, ff2, g2, be2, out, nullptr);
}
